// matchGAT_2353642078853
// MI455X (gfx1250) — compile-verified, matrix operand order checked by reading, not run on hardware
//
#include <hip/hip_runtime.h>
#include <hip/hip_bf16.h>

typedef __attribute__((ext_vector_type(16))) _Float16 v16h;
typedef __attribute__((ext_vector_type(8)))  float    v8f;
typedef unsigned int u32x4 __attribute__((ext_vector_type(4)));
typedef int          i32x4 __attribute__((ext_vector_type(4)));
typedef int          i32x8 __attribute__((ext_vector_type(8)));

#define NN   4096      // nodes per graph
#define NT   8192      // total rows (2 graphs)
#define CC   128       // channels
#define C2   256       // 2*C

#if defined(__HIP_DEVICE_COMPILE__) && __has_builtin(__builtin_amdgcn_tensor_load_to_lds)
#define HAS_TDM 1
#else
#define HAS_TDM 0
#endif

// ---------------------------------------------------------------------------
// Fragment index helper: k offset within a 32-deep K tile for (lane, half h)
//   k = (h&7) + ((h>>3)<<4) + ((lane>>4)<<3)
// ---------------------------------------------------------------------------

// Pack f32 row-major (M x K) into WMMA A-fragment order, f16.
__global__ void pack_a16_k(const float* __restrict__ src, _Float16* __restrict__ dst,
                           int K) {
  int idx  = blockIdx.x * 256 + threadIdx.x;
  int h    = idx & 15;
  int lane = (idx >> 4) & 31;
  int rest = idx >> 9;               // mtile*KT + kt
  int KT   = K >> 5;
  int kt    = rest % KT;
  int mtile = rest / KT;
  int row = mtile * 16 + (lane & 15);
  int k   = kt * 32 + (h & 7) + ((h >> 3) << 4) + ((lane >> 4) << 3);
  dst[idx] = (_Float16)src[row * K + k];
}

// Pack into WMMA B-fragment order (B is K x N logical), f16.
// trans==0: src is (K x N) row-major.  trans==1: src is (N x K) row-major.
__global__ void pack_b16_k(const float* __restrict__ src, _Float16* __restrict__ dst,
                           int K, int N, int trans) {
  int idx  = blockIdx.x * 256 + threadIdx.x;
  int h    = idx & 15;
  int lane = (idx >> 4) & 31;
  int rest = idx >> 9;               // ntile*KT + kt
  int KT   = K >> 5;
  int kt    = rest % KT;
  int ntile = rest / KT;
  int col = ntile * 16 + (lane & 15);
  int k   = kt * 32 + (h & 7) + ((h >> 3) << 4) + ((lane >> 4) << 3);
  float v = trans ? src[col * K + k] : src[k * N + col];
  dst[idx] = (_Float16)v;
}

// ---------------------------------------------------------------------------
// p = W^T a ; c = b . a   (W: 128x128 row-major, a,b: 128)
__global__ void vec_proj_k(const float* __restrict__ W, const float* __restrict__ b,
                           const float* __restrict__ a, float* __restrict__ p,
                           float* __restrict__ c) {
  __shared__ float s[CC];
  int t = threadIdx.x;
  float acc = 0.f;
  for (int i = 0; i < CC; ++i) acc += W[i * CC + t] * a[i];
  p[t] = acc;
  s[t] = b[t] * a[t];
  __syncthreads();
  for (int st = 64; st > 0; st >>= 1) {
    if (t < st) s[t] += s[t + st];
    __syncthreads();
  }
  if (t == 0) *c = s[0];
}

// out[r] = X[r,:] . p + c   (one 128-thread block per row)
__global__ void rowdot_k(const float* __restrict__ X, const float* __restrict__ p,
                         const float* __restrict__ c, float* __restrict__ out) {
  __shared__ float s[CC];
  int r = blockIdx.x, t = threadIdx.x;
  s[t] = X[r * CC + t] * p[t];
  __syncthreads();
  for (int st = 64; st > 0; st >>= 1) {
    if (t < st) s[t] += s[t + st];
    __syncthreads();
  }
  if (t == 0) out[r] = s[0] + *c;
}

__global__ void reduce_max_k(const float* __restrict__ x, float* __restrict__ out, int n) {
  __shared__ float s[256];
  float m = -1e30f;
  for (int i = threadIdx.x; i < n; i += 256) m = fmaxf(m, x[i]);
  s[threadIdx.x] = m;
  __syncthreads();
  for (int st = 128; st > 0; st >>= 1) {
    if (threadIdx.x < st) s[threadIdx.x] = fmaxf(s[threadIdx.x], s[threadIdx.x + st]);
    __syncthreads();
  }
  if (threadIdx.x == 0) *out = s[0];
}

// per row: rmax = lrelu(al[r] + maxAr); invZ = 1/sum_m exp(lrelu(al[r]+ar[m]) - rmax)
__global__ void rowstats_k(const float* __restrict__ al, const float* __restrict__ ar,
                           const float* __restrict__ mAr, float* __restrict__ rmx,
                           float* __restrict__ ivz) {
  __shared__ float s[256];
  int r = blockIdx.x, t = threadIdx.x;
  float alr = al[r];
  float rm = alr + *mAr;
  rm = rm > 0.f ? rm : 0.2f * rm;
  float acc = 0.f;
  for (int m = t; m < NT; m += 256) {
    float v = alr + ar[m];
    v = v > 0.f ? v : 0.2f * v;
    acc += __expf(v - rm);
  }
  s[t] = acc;
  __syncthreads();
  for (int st = 128; st > 0; st >>= 1) {
    if (t < st) s[t] += s[t + st];
    __syncthreads();
  }
  if (t == 0) { rmx[r] = rm; ivz[r] = 1.f / s[0]; }
}

// ---------------------------------------------------------------------------
// Attention GEMM: O = softmax_row(lrelu(al+ar)) @ tgt, fused into
// feat[r, 0:128] = X[r], feat[r, 128:256] = X[r] - O[r].
// M = NT, K = NT, N = 128.  A fragments generated on the fly (exp in VALU),
// a_r staged in 32 KB of LDS via the Tensor Data Mover when available.
__global__ __launch_bounds__(128) void att_gemm_k(
    const float* __restrict__ al, const float* __restrict__ ar,
    const float* __restrict__ rmx, const float* __restrict__ ivz,
    const _Float16* __restrict__ Bp, const float* __restrict__ X,
    float* __restrict__ feat) {
  __shared__ float s_ar[NT];
  int tid = threadIdx.x;
  int wid = tid >> 5, lane = tid & 31;

#if HAS_TDM
  // TDM: one wave DMAs the whole a_r vector (8192 f32 = 32 KB) into LDS as a
  // 1-row tile of a 1-D tensor (data_size=4B, tile_dim0=8192, tile_dim1=1).
  if (wid == 0) {
    unsigned long long ga = (unsigned long long)(uintptr_t)ar;
    unsigned int lds_off = (unsigned int)(uintptr_t)s_ar;   // LDS aperture low bits
    u32x4 g0;
    g0[0] = 1u;                                             // count=1, user descriptor
    g0[1] = lds_off;                                        // lds_addr
    g0[2] = (unsigned int)ga;                               // global_addr[31:0]
    g0[3] = (unsigned int)((ga >> 32) & 0x1FFFFFFu) | (2u << 30);  // [56:32] | type=2
    i32x8 g1;
    g1[0] = (int)(2u << 16);                 // data_size = 2 (4 bytes)
    g1[1] = (int)((NT & 0xFFFF) << 16);      // tensor_dim0[15:0] @ bits 63:48
    g1[2] = (int)((NT >> 16) | (1 << 16));   // tensor_dim0[31:16] | tensor_dim1=1
    g1[3] = (int)((NT & 0xFFFF) << 16);      // tile_dim0 = 8192 @ bits 127:112
    g1[4] = 1;                               // tile_dim1 = 1
    g1[5] = NT;                              // tensor_dim0_stride[31:0]
    g1[6] = 0;
    g1[7] = 0;
    i32x4 gz = {0, 0, 0, 0};
#if __clang_major__ >= 23
    i32x8 gz8 = {0, 0, 0, 0, 0, 0, 0, 0};
    __builtin_amdgcn_tensor_load_to_lds(g0, g1, gz, gz, gz8, 0);
#else
    __builtin_amdgcn_tensor_load_to_lds(g0, g1, gz, gz, 0);
#endif
    __builtin_amdgcn_s_wait_tensorcnt(0);
  }
  __syncthreads();
#else
  for (int i = tid; i < NT; i += 128) s_ar[i] = ar[i];
  __syncthreads();
#endif

  int mtile = blockIdx.x * 4 + wid;
  int row = mtile * 16 + (lane & 15);
  float alr = al[row], rm = rmx[row], iz = ivz[row];
  int koff = (lane >> 4) << 3;

  v8f acc[8] = {};
  const v16h* B = (const v16h*)Bp;
  const int KT = NT / 32;  // 256

  for (int kt = 0; kt < KT; ++kt) {
    v16h a;
    int kbase = kt * 32 + koff;
#pragma unroll
    for (int h = 0; h < 16; ++h) {
      int k = kbase + (h & 7) + ((h >> 3) << 4);
      float sVal = alr + s_ar[k];
      sVal = sVal > 0.f ? sVal : 0.2f * sVal;
      a[h] = (_Float16)(__expf(sVal - rm) * iz);
    }
#pragma unroll
    for (int t = 0; t < 8; ++t) {
      v16h b = B[(t * KT + kt) * 32 + lane];
      acc[t] = __builtin_amdgcn_wmma_f32_16x16x32_f16(
          false, a, false, b, (short)0, acc[t], false, false);
    }
  }

  int rl = lane & 15, hb = (lane >> 4) * 8;
#pragma unroll
  for (int t = 0; t < 8; ++t) {
    int ncol = t * 16 + rl;
#pragma unroll
    for (int r = 0; r < 8; ++r) {
      int rr = mtile * 16 + hb + r;
      float xv = X[rr * CC + ncol];
      feat[rr * C2 + ncol]      = xv;
      feat[rr * C2 + CC + ncol] = xv - acc[t][r];
    }
  }
}

// ---------------------------------------------------------------------------
// Generic packed WMMA GEMM: out (M x N, f32) = Ap x Bp.
// mode 0: out += bias[col].  mode 1: out = sigmoid(acc).
// grid = (M/64, N/128); 4 waves per block, each wave a 16x128 strip.
__global__ __launch_bounds__(128) void gemm_packed_k(
    const _Float16* __restrict__ Ap, const _Float16* __restrict__ Bp,
    const float* __restrict__ bias, float* __restrict__ out,
    int N, int KT, int mode) {
  int tid = threadIdx.x, wid = tid >> 5, lane = tid & 31;
  int mtile = blockIdx.x * 4 + wid;
  int nb = blockIdx.y * 8;
  v8f acc[8] = {};
  const v16h* A = (const v16h*)Ap;
  const v16h* B = (const v16h*)Bp;
  for (int kt = 0; kt < KT; ++kt) {
    v16h a = A[(mtile * KT + kt) * 32 + lane];
#pragma unroll
    for (int t = 0; t < 8; ++t) {
      v16h b = B[((nb + t) * KT + kt) * 32 + lane];
      acc[t] = __builtin_amdgcn_wmma_f32_16x16x32_f16(
          false, a, false, b, (short)0, acc[t], false, false);
    }
  }
  int rl = lane & 15, hb = (lane >> 4) * 8;
#pragma unroll
  for (int t = 0; t < 8; ++t) {
    int ncol = (nb + t) * 16 + rl;
#pragma unroll
    for (int r = 0; r < 8; ++r) {
      int row = mtile * 16 + hb + r;
      float v = acc[t][r];
      if (mode == 0) v += bias[ncol];
      else           v = 1.f / (1.f + __expf(-v));
      out[row * N + ncol] = v;
    }
  }
}

// ---------------------------------------------------------------------------
// Gram GEMM: out = sigmoid(A x B), 32x128 strip per wave (16 accumulators,
// 2 A fragments -> 16 WMMA per 10 fragment loads per K-step, half the B traffic).
// grid = (M/128, N/128); 4 waves per block.
__global__ __launch_bounds__(128) void gemm_gram32_k(
    const _Float16* __restrict__ Ap, const _Float16* __restrict__ Bp,
    float* __restrict__ out, int N, int KT) {
  int tid = threadIdx.x, wid = tid >> 5, lane = tid & 31;
  int mtile0 = (blockIdx.x * 4 + wid) * 2;   // two consecutive 16-row tiles
  int nb = blockIdx.y * 8;
  v8f acc0[8] = {};
  v8f acc1[8] = {};
  const v16h* A = (const v16h*)Ap;
  const v16h* B = (const v16h*)Bp;
  for (int kt = 0; kt < KT; ++kt) {
    v16h a0 = A[(mtile0 * KT + kt) * 32 + lane];
    v16h a1 = A[((mtile0 + 1) * KT + kt) * 32 + lane];
#pragma unroll
    for (int t = 0; t < 8; ++t) {
      v16h b = B[((nb + t) * KT + kt) * 32 + lane];
      acc0[t] = __builtin_amdgcn_wmma_f32_16x16x32_f16(
          false, a0, false, b, (short)0, acc0[t], false, false);
      acc1[t] = __builtin_amdgcn_wmma_f32_16x16x32_f16(
          false, a1, false, b, (short)0, acc1[t], false, false);
    }
  }
  int rl = lane & 15, hb = (lane >> 4) * 8;
#pragma unroll
  for (int t = 0; t < 8; ++t) {
    int ncol = (nb + t) * 16 + rl;
#pragma unroll
    for (int r = 0; r < 8; ++r) {
      int row0 = mtile0 * 16 + hb + r;
      out[row0 * N + ncol]        = 1.f / (1.f + __expf(-acc0[t][r]));
      out[(row0 + 16) * N + ncol] = 1.f / (1.f + __expf(-acc1[t][r]));
    }
  }
}

// ---------------------------------------------------------------------------
__global__ void zero_k(float* __restrict__ p, int n) {
  int i = blockIdx.x * 256 + threadIdx.x;
  if (i < n) p[i] = 0.f;
}

// One block per edge (256 threads = one per feat channel).
__global__ void scatter_k(const int* __restrict__ ei, const float* __restrict__ feat,
                          float* __restrict__ sums, float* __restrict__ cnt,
                          int base, int E) {
  int e = blockIdx.x;
  int s = ei[e] + base;
  int d = ei[E + e] + base;
  atomicAdd(&sums[s * C2 + threadIdx.x], feat[d * C2 + threadIdx.x]);
  if (threadIdx.x == 0) atomicAdd(&cnt[s], 1.0f);
}

// aggr = relu((sums + feat_selfloop) / (cnt + 1))
__global__ void aggr_k(const float* __restrict__ sums, const float* __restrict__ feat,
                       const float* __restrict__ cnt, float* __restrict__ aggr) {
  int i = blockIdx.x * 256 + threadIdx.x;  // NT * C2 total
  int n = i >> 8;
  float v = (sums[i] + feat[i]) / (cnt[n] + 1.0f);
  aggr[i] = v > 0.f ? v : 0.f;
}

// ---------------------------------------------------------------------------
extern "C" void kernel_launch(void* const* d_in, const int* in_sizes, int n_in,
                              void* d_out, int out_size, void* d_ws, size_t ws_size,
                              hipStream_t stream) {
  const float* x1  = (const float*)d_in[0];
  const float* x2  = (const float*)d_in[1];
  const int*   ei1 = (const int*)d_in[2];
  const int*   ei2 = (const int*)d_in[3];
  const int E = in_sizes[2] / 2;

  // per-layer params: wl_w, wl_b, wr_w, wr_b, al, ar, ww, wb
  const float* wlw[2] = {(const float*)d_in[4],  (const float*)d_in[10]};
  const float* wlb[2] = {(const float*)d_in[5],  (const float*)d_in[11]};
  const float* wrw[2] = {(const float*)d_in[6],  (const float*)d_in[12]};
  const float* wrb[2] = {(const float*)d_in[7],  (const float*)d_in[13]};
  const float* alv[2] = {(const float*)d_in[8],  (const float*)d_in[14]};
  const float* arv[2] = {(const float*)d_in[9],  (const float*)d_in[15]};
  const float* www[2] = {(const float*)d_in[16], (const float*)d_in[18]};
  const float* wwb[2] = {(const float*)d_in[17], (const float*)d_in[19]};
  const float* linw = (const float*)d_in[20];
  const float* linb = (const float*)d_in[21];

  // ---- workspace carve ----
  size_t off = 0;
  auto carve = [&](size_t bytes) -> void* {
    void* p = (char*)d_ws + off;
    off += (bytes + 255) & ~(size_t)255;
    return p;
  };
  float* Xbuf  = (float*)carve((size_t)NT * CC * 4);
  float* Xbuf2 = (float*)carve((size_t)NT * CC * 4);
  float* feat  = (float*)carve((size_t)NT * C2 * 4);
  float* sums  = (float*)carve((size_t)NT * C2 * 4);
  float* aggr  = (float*)carve((size_t)NT * C2 * 4);
  float* feats = (float*)carve((size_t)NT * CC * 4);
  float* cnt   = (float*)carve((size_t)NT * 4);
  float* a_l   = (float*)carve((size_t)NT * 4);
  float* a_r   = (float*)carve((size_t)NT * 4);
  float* rmx   = (float*)carve((size_t)NT * 4);
  float* ivz   = (float*)carve((size_t)NT * 4);
  float* p_l   = (float*)carve(CC * 4);
  float* c_l   = (float*)carve(4);
  float* p_r   = (float*)carve(CC * 4);
  float* c_r   = (float*)carve(4);
  float* mAr   = (float*)carve(4);
  _Float16* tgtP  = (_Float16*)carve((size_t)NT * CC * 2);
  _Float16* aggrP = (_Float16*)carve((size_t)NT * C2 * 2);
  _Float16* wwP   = (_Float16*)carve((size_t)C2 * CC * 2);
  _Float16* XP    = (_Float16*)carve((size_t)NT * CC * 2);   // reused as feats-A pack
  _Float16* linP  = (_Float16*)carve((size_t)CC * CC * 2);
  _Float16* fBp   = (_Float16*)carve((size_t)CC * NT * 2);

  // initial X = concat(x1, x2)
  hipMemcpyAsync(Xbuf,            x1, (size_t)NN * CC * 4, hipMemcpyDeviceToDevice, stream);
  hipMemcpyAsync(Xbuf + NN * CC,  x2, (size_t)NN * CC * 4, hipMemcpyDeviceToDevice, stream);

  float* X = Xbuf;
  float* Xn = Xbuf2;

  for (int l = 0; l < 2; ++l) {
    // alpha projections: a_l = X @ (wl^T al) + bl.al ; a_r = X @ (wr^T ar) + br.ar
    vec_proj_k<<<1, 128, 0, stream>>>(wlw[l], wlb[l], alv[l], p_l, c_l);
    vec_proj_k<<<1, 128, 0, stream>>>(wrw[l], wrb[l], arv[l], p_r, c_r);
    rowdot_k<<<NT, 128, 0, stream>>>(X, p_l, c_l, a_l);
    rowdot_k<<<NT, 128, 0, stream>>>(X, p_r, c_r, a_r);
    reduce_max_k<<<1, 256, 0, stream>>>(a_r, mAr, NT);
    rowstats_k<<<NT, 256, 0, stream>>>(a_l, a_r, mAr, rmx, ivz);

    // pack tgt (= X, K=NT rows x N=CC cols) as B fragments
    pack_b16_k<<<(NT * CC) / 256, 256, 0, stream>>>(X, tgtP, NT, CC, 0);

    // fused attention GEMM -> feat = [X, X - att@tgt]
    att_gemm_k<<<NT / 64, 128, 0, stream>>>(a_l, a_r, rmx, ivz, tgtP, X, feat);

    // scatter-mean over edges (+self-loop folded into aggr_k)
    zero_k<<<(NT * C2) / 256, 256, 0, stream>>>(sums, NT * C2);
    zero_k<<<(NT + 255) / 256, 256, 0, stream>>>(cnt, NT);
    scatter_k<<<E, 256, 0, stream>>>(ei1, feat, sums, cnt, 0, E);
    scatter_k<<<E, 256, 0, stream>>>(ei2, feat, sums, cnt, NN, E);
    aggr_k<<<(NT * C2) / 256, 256, 0, stream>>>(sums, feat, cnt, aggr);

    // update: Xn = relu(aggr) @ ww^T + wb   (M=NT, K=C2, N=CC)
    pack_a16_k<<<(NT * C2) / 256, 256, 0, stream>>>(aggr, aggrP, C2);
    pack_b16_k<<<(C2 * CC) / 256, 256, 0, stream>>>(www[l], wwP, C2, CC, 1);
    gemm_packed_k<<<dim3(NT / 64, 1), 128, 0, stream>>>(aggrP, wwP, wwb[l], Xn, CC, C2 / 32, 0);

    float* tmp = X; X = Xn; Xn = tmp;
  }

  // feats = X @ lin_w^T + lin_b   (M=NT, K=CC, N=CC)
  pack_a16_k<<<(NT * CC) / 256, 256, 0, stream>>>(X, XP, CC);
  pack_b16_k<<<(CC * CC) / 256, 256, 0, stream>>>(linw, linP, CC, CC, 1);
  gemm_packed_k<<<dim3(NT / 64, 1), 128, 0, stream>>>(XP, linP, linb, feats, CC, CC / 32, 0);

  // out = sigmoid(feats @ feats^T)   (M=NT, K=CC, N=NT), 32-row strips per wave
  pack_a16_k<<<(NT * CC) / 256, 256, 0, stream>>>(feats, XP, CC);
  pack_b16_k<<<(NT * CC) / 256, 256, 0, stream>>>(feats, fBp, CC, NT, 1);
  gemm_gram32_k<<<dim3(NT / 128, NT / 128), 128, 0, stream>>>(
      XP, fBp, (float*)d_out, NT, CC / 32);
}
